// GPSDepth_2147483648108
// MI455X (gfx1250) — matrix operand order repaired and checked by reading, not compiled, not run on hardware
//
#include <hip/hip_runtime.h>
#include <math.h>

// MI455X / gfx1250 implementation of the 3-layer graph-attention network.
// Dense GEMMs: pre-converted bf16 operands (H row-major, W transposed) so each
// 16x16x32 WMMA step is fed by 4x global_load_b128, then
// v_wmma_f32_16x16x32_bf16 with f32 accumulation.
// Edge aggregation: per-wave float4 gathers + global f32 atomic scatters
// (workload is HBM/atomic bound: ~1.6GB gather + 410M atomics per wide layer).
//
// Workspace requirement: ~280 MB.

#define ALPHA 0.2f
#define BN_EPS 1e-5f

typedef __attribute__((ext_vector_type(16))) __bf16 v16bf;
typedef __attribute__((ext_vector_type(8)))  __bf16 v8bf;   // 16 bytes -> b128
typedef __attribute__((ext_vector_type(8)))  float  v8f;

__device__ __forceinline__ float wave_sum(float v) {
#pragma unroll
  for (int o = 16; o > 0; o >>= 1) v += __shfl_xor(v, o, 32);
  return v;
}

// ---------------------------------------------------------------------------
// f32 -> bf16 conversion (row-major copy)
// ---------------------------------------------------------------------------
__global__ void k_cvt_bf16(const float* __restrict__ src, __bf16* __restrict__ dst,
                           long long n)
{
  const long long stride = (long long)gridDim.x * blockDim.x;
  for (long long i = (long long)blockIdx.x * blockDim.x + threadIdx.x; i < n; i += stride)
    dst[i] = (__bf16)src[i];
}

// W[K,Nc] (f32) -> WT[Nc,K] (bf16)
__global__ void k_cvt_w_t(const float* __restrict__ W, __bf16* __restrict__ WT,
                          int K, int Nc)
{
  const int idx = blockIdx.x * blockDim.x + threadIdx.x;
  if (idx >= K * Nc) return;
  const int k = idx / Nc;
  const int n = idx % Nc;
  WT[(size_t)n * K + k] = (__bf16)W[idx];
}

// ---------------------------------------------------------------------------
// NH[M,Nc] = HB[M,K] @ WT[Nc,K]^T + Bias[Nc]   (bf16 WMMA, f32 accumulate)
// One wave computes one 16x16 output tile; K-loop fully unrolled.
// Fragment layouts follow CDNA5 ISA 7.12.2:
//  A (16x32 bf16): lane m=(lane&15), half=lane>>4; VGPR0-3 = K[k0+8h .. +8),
//                  VGPR4-7 = K[k0+16+8h .. +8)        -> 2x b128 from HB row
//  B (32x16 bf16): lane n=(lane&15); VGPR0-3 = K[k0+8h .. +8),
//                  VGPR4-7 = K[k0+16+8h .. +8)        -> 2x b128 from WT row
//  C/D (f32): VGPR j -> row (j + 8*half), col (lane&15)
// ---------------------------------------------------------------------------
template <int K>
__global__ __launch_bounds__(256) void k_gemm_bf16(
    const __bf16* __restrict__ HB, const __bf16* __restrict__ WT,
    const float* __restrict__ Bias, float* __restrict__ NH, int M, int Nc)
{
  const int lane   = threadIdx.x & 31;
  const int wave   = threadIdx.x >> 5;
  const int tile_n = blockIdx.x;
  const int tile_m = blockIdx.y * 8 + wave;
  const int mtiles = M >> 4;
  if (tile_m >= mtiles) return;            // wave-uniform: EXEC stays all-ones

  const int l16  = lane & 15;
  const int half = lane >> 4;
  const int m = (tile_m << 4) + l16;
  const int n = (tile_n << 4) + l16;
  const __bf16* __restrict__ arow = HB + (size_t)m * K;
  const __bf16* __restrict__ brow = WT + (size_t)n * K;

  v8f acc = {0.f, 0.f, 0.f, 0.f, 0.f, 0.f, 0.f, 0.f};

#pragma unroll
  for (int k0 = 0; k0 < K; k0 += 32) {
    const v8bf alo = *(const v8bf*)(arow + k0 + (half << 3));
    const v8bf ahi = *(const v8bf*)(arow + k0 + 16 + (half << 3));
    const v8bf blo = *(const v8bf*)(brow + k0 + (half << 3));
    const v8bf bhi = *(const v8bf*)(brow + k0 + 16 + (half << 3));
    const v16bf afrag = __builtin_shufflevector(
        alo, ahi, 0, 1, 2, 3, 4, 5, 6, 7, 8, 9, 10, 11, 12, 13, 14, 15);
    const v16bf bfrag = __builtin_shufflevector(
        blo, bhi, 0, 1, 2, 3, 4, 5, 6, 7, 8, 9, 10, 11, 12, 13, 14, 15);
    acc = __builtin_amdgcn_wmma_f32_16x16x32_bf16(
        /*neg_a=*/false, afrag, /*neg_b=*/false, bfrag,
        /*c_mod=*/(short)0, acc, /*reuse_a=*/false, /*reuse_b=*/false);
  }

  const float b = Bias[n];
#pragma unroll
  for (int j = 0; j < 8; ++j) {
    const int r = (tile_m << 4) + j + (half << 3);
    NH[(size_t)r * Nc + n] = acc[j] + b;
  }
}

// ---------------------------------------------------------------------------
// a_la[i] = NH[i,:].la + Bla ; a_ra[i] = NH[i,:].ra + Bra   (wave per node)
// ---------------------------------------------------------------------------
__global__ __launch_bounds__(256) void k_att_vec(
    const float* __restrict__ NH, const float* __restrict__ la,
    const float* __restrict__ ra, const float* __restrict__ Bla,
    const float* __restrict__ Bra, float* __restrict__ a_la,
    float* __restrict__ a_ra, int M, int Nc)
{
  const int node = blockIdx.x * 8 + (threadIdx.x >> 5);
  const int lane = threadIdx.x & 31;
  if (node >= M) return;
  const float* row = NH + (size_t)node * Nc;
  float sl = 0.f, sr = 0.f;
  for (int k = lane; k < Nc; k += 32) {
    const float v = row[k];
    sl += v * la[k];
    sr += v * ra[k];
  }
  sl = wave_sum(sl);
  sr = wave_sum(sr);
  if (lane == 0) {
    a_la[node] = sl + Bla[0];
    a_ra[node] = sr + Bra[0];
  }
}

// ---------------------------------------------------------------------------
// a_edge = exp(-leakyrelu((a_la[src]+a_ra[dst])*inv_sqrt_f)); rowsum scatter
// ---------------------------------------------------------------------------
__global__ void k_edge_att(const int* __restrict__ src, const int* __restrict__ dst,
                           const float* __restrict__ a_la, const float* __restrict__ a_ra,
                           float* __restrict__ a_edge, float* __restrict__ row_sum,
                           float inv_sqrt_f, int E)
{
  const int e = blockIdx.x * blockDim.x + threadIdx.x;
  if (e >= E) return;
  const int s = src[e], d = dst[e];
  const float x  = (a_la[s] + a_ra[d]) * inv_sqrt_f;
  const float lr = x > 0.f ? x : ALPHA * x;
  const float ae = __expf(-lr);
  a_edge[e] = ae;
  atomicAdd(&row_sum[s], ae);
}

// ---------------------------------------------------------------------------
// AGG[src,:] += a_edge * NH[dst,:]   (wave per edge, float4 gathers)
// ---------------------------------------------------------------------------
__global__ __launch_bounds__(256) void k_agg(
    const int* __restrict__ src, const int* __restrict__ dst,
    const float* __restrict__ a_edge, const float* __restrict__ NH,
    float* __restrict__ AGG, int E, int Nc)
{
  const int e = blockIdx.x * 8 + (threadIdx.x >> 5);
  const int lane = threadIdx.x & 31;
  if (e >= E) return;
  const int s = src[e], d = dst[e];
  const float ae = a_edge[e];
  const float4* __restrict__ row4 = (const float4*)(NH + (size_t)d * Nc);
  float* __restrict__ out = AGG + (size_t)s * Nc;
  const int n4 = Nc >> 2;
  for (int k4 = lane; k4 < n4; k4 += 32) {
    const float4 v = row4[k4];
    const int k = k4 << 2;
    atomicAdd(&out[k + 0], ae * v.x);
    atomicAdd(&out[k + 1], ae * v.y);
    atomicAdd(&out[k + 2], ae * v.z);
    atomicAdd(&out[k + 3], ae * v.w);
  }
}

// ---------------------------------------------------------------------------
// FH = aggr*FH/max(rowsum,1e-16) + (1-aggr)*NH    (FH holds agg on input)
// ---------------------------------------------------------------------------
__global__ void k_blend(const float* __restrict__ NH, float* __restrict__ FH,
                        const float* __restrict__ rowsum, const float* __restrict__ aggr,
                        int M, int Nc)
{
  const long long idx = (long long)blockIdx.x * blockDim.x + threadIdx.x;
  const long long tot = (long long)M * Nc;
  if (idx >= tot) return;
  const int i = (int)(idx / Nc);
  const float rs = fmaxf(rowsum[i], 1e-16f);
  const float a  = aggr[i];
  FH[idx] = a * (FH[idx] / rs) + (1.f - a) * NH[idx];
}

// ---------------------------------------------------------------------------
// per-column sum & sumsq (block = Nc threads, chunked rows -> few atomics)
// ---------------------------------------------------------------------------
__global__ void k_colstats(const float* __restrict__ FH, float* __restrict__ ssum,
                           float* __restrict__ ssq, int M, int Nc)
{
  const int c  = threadIdx.x;
  const int r0 = blockIdx.x * 128;
  const int r1 = min(r0 + 128, M);
  float s = 0.f, q = 0.f;
  for (int r = r0; r < r1; ++r) {
    const float v = FH[(size_t)r * Nc + c];
    s += v;
    q += v * v;
  }
  atomicAdd(&ssum[c], s);
  atomicAdd(&ssq[c], q);
}

// BatchNorm (biased var, training stats) + ReLU
__global__ void k_bn_relu(float* __restrict__ FH, const float* __restrict__ ssum,
                          const float* __restrict__ ssq, const float* __restrict__ g,
                          const float* __restrict__ b, int M, int Nc)
{
  const long long idx = (long long)blockIdx.x * blockDim.x + threadIdx.x;
  const long long tot = (long long)M * Nc;
  if (idx >= tot) return;
  const int c = (int)(idx % Nc);
  const float invM = 1.f / (float)M;
  const float mu  = ssum[c] * invM;
  const float var = ssq[c] * invM - mu * mu;
  const float v = (FH[idx] - mu) * rsqrtf(var + BN_EPS) * g[c] + b[c];
  FH[idx] = fmaxf(v, 0.f);
}

// ---------------------------------------------------------------------------
// HM[i,0:16] = FH[i,:] @ W2 + B2     (16 threads per node)
// ---------------------------------------------------------------------------
__global__ void k_hmini(const float* __restrict__ FH, const float* __restrict__ W2,
                        const float* __restrict__ B2, float* __restrict__ HM,
                        int M, int Nc)
{
  const int t = blockIdx.x * blockDim.x + threadIdx.x;
  const int i = t >> 4;
  const int j = t & 15;
  if (i >= M) return;
  const float* row = FH + (size_t)i * Nc;
  float s = B2[j];
  for (int k = 0; k < Nc; ++k) s += row[k] * W2[k * 16 + j];
  HM[(size_t)i * 16 + j] = s;
}

// ---------------------------------------------------------------------------
// edge MLP: f = sigmoid(lf2(tanh(lf1([hs,hd,|hd-hs|])))); fsum scatter
// lf1_w is (16,48) row-major; lf2_w is (1,16).
// ---------------------------------------------------------------------------
__global__ __launch_bounds__(256) void k_edge_mlp(
    const int* __restrict__ src, const int* __restrict__ dst,
    const float* __restrict__ HM, const float* __restrict__ lf1w,
    const float* __restrict__ lf1b, const float* __restrict__ lf2w,
    const float* __restrict__ lf2b, float* __restrict__ fe,
    float* __restrict__ fsum, int E)
{
  __shared__ float w1[16 * 48];
  __shared__ float b1[16];
  __shared__ float w2[16];
  for (int t = threadIdx.x; t < 16 * 48; t += blockDim.x) w1[t] = lf1w[t];
  if (threadIdx.x < 16) {
    b1[threadIdx.x] = lf1b[threadIdx.x];
    w2[threadIdx.x] = lf2w[threadIdx.x];
  }
  __syncthreads();

  const int e = blockIdx.x * blockDim.x + threadIdx.x;
  if (e >= E) return;
  const int s = src[e], d = dst[e];
  float feat[48];
#pragma unroll
  for (int k = 0; k < 16; ++k) {
    const float hs = HM[(size_t)s * 16 + k];
    const float hd = HM[(size_t)d * 16 + k];
    feat[k]      = hs;
    feat[16 + k] = hd;
    feat[32 + k] = fabsf(hd - hs);
  }
  float out = lf2b[0];
#pragma unroll
  for (int j = 0; j < 16; ++j) {
    float a = b1[j];
#pragma unroll
    for (int k = 0; k < 48; ++k) a += feat[k] * w1[j * 48 + k];
    out += tanhf(a) * w2[j];
  }
  const float f = 1.f / (1.f + __expf(-out));
  fe[e] = f;
  atomicAdd(&fsum[s], f);
}

// res1 = fsum / max(deg,1)   (in place)
__global__ void k_div_deg(float* __restrict__ v, const float* __restrict__ deg, int M)
{
  const int i = blockIdx.x * blockDim.x + threadIdx.x;
  if (i >= M) return;
  v[i] = v[i] / fmaxf(deg[i], 1.f);
}

// r2sum[src] += f[e] * res1[dst]
__global__ void k_res2_accum(const int* __restrict__ src, const int* __restrict__ dst,
                             const float* __restrict__ fe, const float* __restrict__ res1,
                             float* __restrict__ r2sum, int E)
{
  const int e = blockIdx.x * blockDim.x + threadIdx.x;
  if (e >= E) return;
  atomicAdd(&r2sum[src[e]], fe[e] * res1[dst[e]]);
}

// aggr *= r2sum / max(deg,1)
__global__ void k_res2_aggr(const float* __restrict__ r2sum, const float* __restrict__ deg,
                            float* __restrict__ aggr, int M)
{
  const int i = blockIdx.x * blockDim.x + threadIdx.x;
  if (i >= M) return;
  aggr[i] *= r2sum[i] / fmaxf(deg[i], 1.f);
}

// log_softmax over 64 classes (wave per node, 2 cols/lane)
__global__ __launch_bounds__(256) void k_logsoftmax(const float* __restrict__ H,
                                                    float* __restrict__ out, int M)
{
  const int node = blockIdx.x * 8 + (threadIdx.x >> 5);
  const int lane = threadIdx.x & 31;
  if (node >= M) return;
  const float* row = H + (size_t)node * 64;
  const float v0 = row[lane], v1 = row[lane + 32];
  float mx = fmaxf(v0, v1);
#pragma unroll
  for (int o = 16; o > 0; o >>= 1) mx = fmaxf(mx, __shfl_xor(mx, o, 32));
  float se = __expf(v0 - mx) + __expf(v1 - mx);
  se = wave_sum(se);
  const float lse = mx + __logf(se);
  out[(size_t)node * 64 + lane]      = v0 - lse;
  out[(size_t)node * 64 + lane + 32] = v1 - lse;
}

// utility
__global__ void k_fill(float* __restrict__ p, float v, long long n)
{
  const long long stride = (long long)gridDim.x * blockDim.x;
  for (long long i = (long long)blockIdx.x * blockDim.x + threadIdx.x; i < n; i += stride)
    p[i] = v;
}

__global__ void k_deg(const int* __restrict__ src, float* __restrict__ deg, int E)
{
  const int e = blockIdx.x * blockDim.x + threadIdx.x;
  if (e >= E) return;
  atomicAdd(&deg[src[e]], 1.f);
}

// ---------------------------------------------------------------------------
struct LayerP {
  const float *W, *B, *la, *ra, *Bla, *Bra, *W2, *B2, *lf1w, *lf1b, *lf2w, *lf2b, *bng, *bnb;
};

static inline int cdiv(long long a, long long b) { return (int)((a + b - 1) / b); }

extern "C" void kernel_launch(void* const* d_in, const int* in_sizes, int n_in,
                              void* d_out, int out_size, void* d_ws, size_t ws_size,
                              hipStream_t stream)
{
  (void)n_in; (void)out_size; (void)ws_size;
  const int N = in_sizes[0] / 128;
  const int E = in_sizes[2] / 2;

  const float* x     = (const float*)d_in[0];
  const int*   edges = (const int*)d_in[2];
  const int*   src   = edges;
  const int*   dst   = edges + E;

  // params: per layer {W,B,la,ra,Bla,Bra,W2,B2,lf1w,lf1b,lf2w,lf2b[,bn_g,bn_b]}
  LayerP L[3];
  int idx = 3;
  for (int l = 0; l < 3; ++l) {
    LayerP& p = L[l];
    p.W    = (const float*)d_in[idx++];
    p.B    = (const float*)d_in[idx++];
    p.la   = (const float*)d_in[idx++];
    p.ra   = (const float*)d_in[idx++];
    p.Bla  = (const float*)d_in[idx++];
    p.Bra  = (const float*)d_in[idx++];
    p.W2   = (const float*)d_in[idx++];
    p.B2   = (const float*)d_in[idx++];
    p.lf1w = (const float*)d_in[idx++];
    p.lf1b = (const float*)d_in[idx++];
    p.lf2w = (const float*)d_in[idx++];
    p.lf2b = (const float*)d_in[idx++];
    if (l < 2) {
      p.bng = (const float*)d_in[idx++];
      p.bnb = (const float*)d_in[idx++];
    } else {
      p.bng = nullptr;
      p.bnb = nullptr;
    }
  }

  // workspace carve-up (f32 slots): ~280 MB
  float* ws = (float*)d_ws;
  size_t o = 0;
  float* BUF0   = ws + o; o += (size_t)N * 256;  // new_h
  float* BUF1   = ws + o; o += (size_t)N * 256;  // agg -> final_h -> next h
  float* A_LA   = ws + o; o += N;
  float* A_RA   = ws + o; o += N;
  float* ROWSUM = ws + o; o += N;
  float* AEDGE  = ws + o; o += E;
  float* HMINI  = ws + o; o += (size_t)N * 16;
  float* FEDGE  = ws + o; o += E;
  float* DEG    = ws + o; o += N;
  float* FSUM   = ws + o; o += N;               // becomes res1 in place
  float* R2SUM  = ws + o; o += N;
  float* AGGR   = ws + o; o += N;
  float* SSUM   = ws + o; o += 256;             // SSUM,SSQ contiguous
  float* SSQ    = ws + o; o += 256;
  __bf16* HB = (__bf16*)(ws + o); o += (size_t)N * 128;  // N*256 bf16 slots
  __bf16* WT = (__bf16*)(ws + o); o += 32768 + 64;       // 256*256 bf16

  // graph degree (src doesn't change across layers) + aggr = 1
  k_fill<<<1024, 256, 0, stream>>>(DEG, 0.f, (long long)N);
  k_deg<<<cdiv(E, 256), 256, 0, stream>>>(src, DEG, E);
  k_fill<<<1024, 256, 0, stream>>>(AGGR, 1.f, (long long)N);

  const float* h_in = x;
  int fin = 128;
  for (int l = 0; l < 3; ++l) {
    const LayerP& P = L[l];
    const int fout = (l < 2) ? 256 : 64;
    float* NH = BUF0;
    float* FH = BUF1;   // h_in (==BUF1 for l>0) is dead after the GEMM

    // one-time bf16 conversion of this layer's operands
    k_cvt_bf16<<<4096, 256, 0, stream>>>(h_in, HB, (long long)N * fin);
    k_cvt_w_t<<<cdiv((long long)fin * fout, 256), 256, 0, stream>>>(P.W, WT, fin, fout);

    dim3 gg(fout / 16, cdiv(N / 16, 8));
    if (fin == 128)
      k_gemm_bf16<128><<<gg, 256, 0, stream>>>(HB, WT, P.B, NH, N, fout);
    else
      k_gemm_bf16<256><<<gg, 256, 0, stream>>>(HB, WT, P.B, NH, N, fout);

    k_att_vec<<<cdiv(N, 8), 256, 0, stream>>>(NH, P.la, P.ra, P.Bla, P.Bra,
                                              A_LA, A_RA, N, fout);
    k_fill<<<1024, 256, 0, stream>>>(ROWSUM, 0.f, (long long)N);
    k_edge_att<<<cdiv(E, 256), 256, 0, stream>>>(src, dst, A_LA, A_RA, AEDGE, ROWSUM,
                                                 1.f / sqrtf((float)fout), E);

    k_fill<<<4096, 256, 0, stream>>>(FH, 0.f, (long long)N * fout);
    k_agg<<<cdiv(E, 8), 256, 0, stream>>>(src, dst, AEDGE, NH, FH, E, fout);
    k_blend<<<cdiv((long long)N * fout, 256), 256, 0, stream>>>(NH, FH, ROWSUM, AGGR, N, fout);

    if (l < 2) {
      // BatchNorm (training stats) + ReLU
      k_fill<<<1, 256, 0, stream>>>(SSUM, 0.f, 512);  // zeros SSUM and SSQ
      k_colstats<<<cdiv(N, 128), fout, 0, stream>>>(FH, SSUM, SSQ, N, fout);
      k_bn_relu<<<cdiv((long long)N * fout, 256), 256, 0, stream>>>(FH, SSUM, SSQ,
                                                                    P.bng, P.bnb, N, fout);
      // diffusion factors feed the NEXT layer's aggr; skip after last layer
      k_hmini<<<cdiv((long long)N * 16, 256), 256, 0, stream>>>(FH, P.W2, P.B2, HMINI, N, fout);
      k_fill<<<1024, 256, 0, stream>>>(FSUM, 0.f, (long long)N);
      k_edge_mlp<<<cdiv(E, 256), 256, 0, stream>>>(src, dst, HMINI, P.lf1w, P.lf1b,
                                                   P.lf2w, P.lf2b, FEDGE, FSUM, E);
      k_div_deg<<<cdiv(N, 256), 256, 0, stream>>>(FSUM, DEG, N);  // FSUM := res1
      k_fill<<<1024, 256, 0, stream>>>(R2SUM, 0.f, (long long)N);
      k_res2_accum<<<cdiv(E, 256), 256, 0, stream>>>(src, dst, FEDGE, FSUM, R2SUM, E);
      k_res2_aggr<<<cdiv(N, 256), 256, 0, stream>>>(R2SUM, DEG, AGGR, N);
    }
    h_in = FH;
    fin = fout;
  }

  k_logsoftmax<<<cdiv(N, 8), 256, 0, stream>>>(BUF1, (float*)d_out, N);
}
